// PositionAwareTopKPooling_12610023981511
// MI455X (gfx1250) — compile-verified, matrix operand order checked by reading, not run on hardware
//
#include <hip/hip_runtime.h>
#include <hip/hip_bf16.h>
#include <math.h>

typedef __bf16 bf16;
typedef __attribute__((ext_vector_type(16))) __bf16 v16bf;
typedef __attribute__((ext_vector_type(8)))  __bf16 v8bf;
typedef __attribute__((ext_vector_type(8)))  float  v8f;

constexpr int B_ = 128, L_ = 2048, D_ = 128, H_ = 256, O_ = 128, K_ = 64;
constexpr float MASK_FILL_ = -1e9f;
constexpr float LN_EPS_ = 1e-5f;

// ---------------------------------------------------------------------------
// Fragment loader for 16-bit WMMA A/B operands (16x32 MxK layout, wave32):
// lane l: row = (l&15), half = l>>4. Lane holds two runs of 8 contiguous K
// values: [k0 + half*8, +8) and [k0 + 16 + half*8, +8). Each run = one 16B load.
// ---------------------------------------------------------------------------
__device__ __forceinline__ v16bf load_frag(const bf16* __restrict__ rowbase,
                                           int stride, int k0) {
  const int lane = threadIdx.x & 31;
  const bf16* p = rowbase + (lane & 15) * stride + k0 + ((lane >> 4) << 3);
  v8bf lo = *(const v8bf*)(p);
  v8bf hi = *(const v8bf*)(p + 16);
  v16bf f;
#pragma unroll
  for (int i = 0; i < 8; ++i) { f[i] = lo[i]; f[8 + i] = hi[i]; }
  return f;
}

__device__ __forceinline__ v8f wmma_bf16(v16bf a, v16bf b, v8f c) {
  return __builtin_amdgcn_wmma_f32_16x16x32_bf16(false, a, false, b,
                                                 (short)0, c, false, false);
}

// ---------------------------------------------------------------------------
// Kernel 0: transpose+convert weights to bf16 W^T (row = output channel).
// ---------------------------------------------------------------------------
constexpr int W1E  = 256 * 384;
constexpr int WE1E = 256 * 256;
constexpr int WE2E = 128 * 256;

__global__ __launch_bounds__(256)
void prepack_kernel(const float* __restrict__ W1, const float* __restrict__ We1,
                    const float* __restrict__ We2, bf16* __restrict__ wt1,
                    bf16* __restrict__ wte1, bf16* __restrict__ wte2) {
  int i = blockIdx.x * 256 + threadIdx.x;
  if (i < W1E) {
    int n = i / 384, k = i - n * 384;
    wt1[i] = (bf16)W1[k * 256 + n];
  } else if (i < W1E + WE1E) {
    int j = i - W1E;
    int n = j >> 8, k = j & 255;
    wte1[j] = (bf16)We1[k * 256 + n];
  } else if (i < W1E + WE1E + WE2E) {
    int j = i - W1E - WE1E;
    int n = j >> 8, k = j & 255;
    wte2[j] = (bf16)We2[k * 128 + n];
  }
}

// ---------------------------------------------------------------------------
// Kernel 1: importance scores. 64 events per 256-thread block (8 waves).
// Each wave owns 2 N-tiles; its 12 B-fragments per N-tile live in registers
// (96 VGPRs) and are reused across 4 M-tiles -> inner loop is ds_load + wmma
// only, no global-load waits on the critical path.
// ---------------------------------------------------------------------------
__global__ __launch_bounds__(256)
void score_kernel(const float* __restrict__ seq, const int* __restrict__ mask,
                  const float* __restrict__ tgt, const float* __restrict__ pos,
                  const bf16* __restrict__ wt1, const float* __restrict__ b1,
                  const float* __restrict__ w2, const float* __restrict__ b2,
                  float* __restrict__ scores) {
  __shared__ __align__(16) bf16 sA[64][392];   // 64 x 384 (+8 pad) ~= 50 KB
  __shared__ float partial[64];
  const int t  = threadIdx.x;
  const int b  = blockIdx.x >> 5;              // L/64 == 32 tiles per batch
  const int l0 = (blockIdx.x & 31) << 6;

  for (int i = t; i < 64 * 384; i += 256) {    // stage x = [seq|pos|tgt] as bf16
    int m = i / 384, k = i - m * 384;
    float v;
    if (k < 128)      v = seq[((size_t)b * L_ + l0 + m) * D_ + k];
    else if (k < 256) v = pos[(size_t)(l0 + m) * D_ + (k - 128)];
    else              v = tgt[b * D_ + (k - 256)];
    sA[m][k] = (bf16)v;
  }
  if (t < 64) partial[t] = b2[0];
  __syncthreads();

  const int wave = t >> 5, lane = t & 31;
  const int half = lane >> 4, nl = lane & 15;

#pragma unroll
  for (int jn = 0; jn < 2; ++jn) {
    const int nt = wave * 2 + jn;              // 8 waves x 2 = 16 N-tiles (H=256)
    v16bf Bf[12];                              // register-resident B fragments
#pragma unroll
    for (int kk = 0; kk < 12; ++kk)
      Bf[kk] = load_frag(wt1 + nt * 16 * 384, 384, kk * 32);
    const int n = nt * 16 + nl;
    const float wn = w2[n], bn = b1[n];

    for (int mt = 0; mt < 4; ++mt) {
      v8f acc = {};
#pragma unroll
      for (int kk = 0; kk < 12; ++kk) {
        v16bf a = load_frag(&sA[mt * 16][0], 392, kk * 32);
        acc = wmma_bf16(a, Bf[kk], acc);
      }
#pragma unroll
      for (int r = 0; r < 8; ++r) {            // C layout: m = r + 8*half
        float h = acc[r] + bn;
        h = h > 0.f ? h : 0.f;
        atomicAdd(&partial[mt * 16 + r + 8 * half], h * wn);   // ds_add_f32
      }
    }
  }
  __syncthreads();
  if (t < 64) {
    int l = l0 + t;
    scores[(size_t)b * L_ + l] =
        (mask[(size_t)b * L_ + l] == 0) ? MASK_FILL_ : partial[t];
  }
}

// ---------------------------------------------------------------------------
// Kernel 2: per-batch top-64 (iterative argmax, ties -> lowest index), then
// sort selected (index, score) pairs ascending by index (chronological).
// ---------------------------------------------------------------------------
__global__ __launch_bounds__(256)
void topk_kernel(const float* __restrict__ scores, int* __restrict__ tidx,
                 float* __restrict__ tval) {
  __shared__ float s[L_];
  __shared__ float redv[256];
  __shared__ int   redi[256];
  __shared__ int   topi[K_];
  __shared__ float topv[K_];
  const int t = threadIdx.x, b = blockIdx.x;

  for (int i = t; i < L_; i += 256) s[i] = scores[(size_t)b * L_ + i];
  __syncthreads();

  for (int it = 0; it < K_; ++it) {
    float bv = -INFINITY; int bi = L_;
    for (int i = t; i < L_; i += 256) {        // ascending i: '>' keeps first tie
      float v = s[i];
      if (v > bv) { bv = v; bi = i; }
    }
    redv[t] = bv; redi[t] = bi;
    __syncthreads();
    for (int off = 128; off > 0; off >>= 1) {
      if (t < off) {
        float ov = redv[t + off]; int oi = redi[t + off];
        if (ov > redv[t] || (ov == redv[t] && oi < redi[t])) {
          redv[t] = ov; redi[t] = oi;
        }
      }
      __syncthreads();
    }
    if (t == 0) { topi[it] = redi[0]; topv[it] = redv[0]; s[redi[0]] = -INFINITY; }
    __syncthreads();
  }

  for (int ph = 0; ph < K_; ++ph) {            // odd-even transposition sort
    int i = 2 * t + (ph & 1);
    if (t < 32 && i + 1 < K_) {
      if (topi[i] > topi[i + 1]) {
        int   ti = topi[i]; topi[i] = topi[i + 1]; topi[i + 1] = ti;
        float tv = topv[i]; topv[i] = topv[i + 1]; topv[i + 1] = tv;
      }
    }
    __syncthreads();
  }
  if (t < K_) { tidx[b * K_ + t] = topi[t]; tval[b * K_ + t] = topv[t]; }
}

// ---------------------------------------------------------------------------
// Kernel 3: gather selected events, encoder MLP (two bf16 WMMA GEMMs with
// register-resident B fragments), softmax-weighted pooling, LayerNorm.
// One 256-thread block per batch row.
// ---------------------------------------------------------------------------
__global__ __launch_bounds__(256)
void encode_pool_kernel(const float* __restrict__ seq, const float* __restrict__ pos,
                        const int* __restrict__ tidx, const float* __restrict__ tval,
                        const bf16* __restrict__ wte1, const float* __restrict__ be1,
                        const bf16* __restrict__ wte2, const float* __restrict__ be2,
                        const float* __restrict__ gam, const float* __restrict__ bet,
                        float* __restrict__ out) {
  __shared__ __align__(16) bf16 sSel[K_][264];   // 64x256 + pad
  __shared__ __align__(16) bf16 sH[K_][264];
  __shared__ float attn[K_];
  __shared__ float pooled[O_];
  __shared__ float stats[2];
  const int t = threadIdx.x, b = blockIdx.x;
  const int wave = t >> 5, lane = t & 31;
  const int half = lane >> 4, nl = lane & 15;

  for (int i = t; i < K_ * 256; i += 256) {      // gather [seq|pos] rows
    int m = i >> 8, k = i & 255;
    int idx = tidx[b * K_ + m];
    float v = (k < 128) ? seq[((size_t)b * L_ + idx) * D_ + k]
                        : pos[(size_t)idx * D_ + (k - 128)];
    sSel[m][k] = (bf16)v;
  }
  if (t < O_) pooled[t] = 0.f;
  if (t == 0) {                                  // softmax over 64 scores
    float mx = -INFINITY;
    for (int k = 0; k < K_; ++k) mx = fmaxf(mx, tval[b * K_ + k]);
    float sum = 0.f;
    for (int k = 0; k < K_; ++k) {
      float e = __expf(tval[b * K_ + k] - mx);
      attn[k] = e; sum += e;
    }
    float inv = 1.f / sum;
    for (int k = 0; k < K_; ++k) attn[k] *= inv;
  }
  __syncthreads();

  // GEMM1: (64x256)@(256x256) + bias, ReLU -> sH. Wave owns 2 N-tiles.
#pragma unroll
  for (int jn = 0; jn < 2; ++jn) {
    const int nt = wave * 2 + jn;
    v16bf Bf[8];
#pragma unroll
    for (int kk = 0; kk < 8; ++kk)
      Bf[kk] = load_frag(wte1 + nt * 16 * 256, 256, kk * 32);
    const int n = nt * 16 + nl;
    const float bn = be1[n];
    for (int mt = 0; mt < 4; ++mt) {
      v8f acc = {};
#pragma unroll
      for (int kk = 0; kk < 8; ++kk) {
        v16bf a = load_frag(&sSel[mt * 16][0], 264, kk * 32);
        acc = wmma_bf16(a, Bf[kk], acc);
      }
#pragma unroll
      for (int r = 0; r < 8; ++r) {
        int m = mt * 16 + r + 8 * half;
        float h = acc[r] + bn;
        sH[m][n] = (bf16)(h > 0.f ? h : 0.f);
      }
    }
  }
  __syncthreads();

  // GEMM2: (64x256)@(256x128); wave owns 1 N-tile; pool over events.
  {
    const int nt = wave;                         // 8 N-tiles (O=128)
    v16bf Bf[8];
#pragma unroll
    for (int kk = 0; kk < 8; ++kk)
      Bf[kk] = load_frag(wte2 + nt * 16 * 256, 256, kk * 32);
    const int n = nt * 16 + nl;
    for (int mt = 0; mt < 4; ++mt) {
      v8f acc = {};
#pragma unroll
      for (int kk = 0; kk < 8; ++kk) {
        v16bf a = load_frag(&sH[mt * 16][0], 264, kk * 32);
        acc = wmma_bf16(a, Bf[kk], acc);
      }
#pragma unroll
      for (int r = 0; r < 8; ++r) {
        int m = mt * 16 + r + 8 * half;
        atomicAdd(&pooled[n], attn[m] * acc[r]);   // ds_add_f32
      }
    }
  }
  __syncthreads();
  if (t < O_) pooled[t] += be2[t];               // sum(attn)==1 -> bias adds once
  __syncthreads();
  if (t == 0) {
    float s1 = 0.f, s2 = 0.f;
    for (int o = 0; o < O_; ++o) { float p = pooled[o]; s1 += p; s2 += p * p; }
    float mu  = s1 * (1.f / O_);
    float var = s2 * (1.f / O_) - mu * mu;
    stats[0] = mu; stats[1] = rsqrtf(var + LN_EPS_);
  }
  __syncthreads();
  if (t < O_)
    out[(size_t)b * O_ + t] = (pooled[t] - stats[0]) * stats[1] * gam[t] + bet[t];
}

// ---------------------------------------------------------------------------
extern "C" void kernel_launch(void* const* d_in, const int* in_sizes, int n_in,
                              void* d_out, int out_size, void* d_ws, size_t ws_size,
                              hipStream_t stream) {
  (void)in_sizes; (void)n_in; (void)out_size; (void)ws_size;
  const float* seq  = (const float*)d_in[0];
  const int*   mask = (const int*)d_in[1];
  const float* tgt  = (const float*)d_in[2];
  const float* pos  = (const float*)d_in[3];
  const float* W1   = (const float*)d_in[4];
  const float* b1   = (const float*)d_in[5];
  const float* W2   = (const float*)d_in[6];
  const float* b2   = (const float*)d_in[7];
  const float* We1  = (const float*)d_in[8];
  const float* be1  = (const float*)d_in[9];
  const float* We2  = (const float*)d_in[10];
  const float* be2  = (const float*)d_in[11];
  const float* gam  = (const float*)d_in[12];
  const float* bet  = (const float*)d_in[13];
  float* out = (float*)d_out;

  char* ws = (char*)d_ws;                       // workspace layout (bytes)
  float* scores  = (float*)(ws);                //     0 .. 1,048,576  (B*L f32)
  int*   top_idx = (int*)  (ws + 1048576);      // + 32 KB (B*64 i32)
  float* top_val = (float*)(ws + 1081344);      // + 32 KB (B*64 f32)
  bf16*  wt1     = (bf16*) (ws + 1114112);      // 256x384 bf16
  bf16*  wte1    = (bf16*) (ws + 1310720);      // 256x256 bf16
  bf16*  wte2    = (bf16*) (ws + 1441792);      // 128x256 bf16

  prepack_kernel<<<(W1E + WE1E + WE2E + 255) / 256, 256, 0, stream>>>(
      W1, We1, We2, wt1, wte1, wte2);
  score_kernel<<<B_ * (L_ / 64), 256, 0, stream>>>(
      seq, mask, tgt, pos, wt1, b1, W2, b2, scores);
  topk_kernel<<<B_, 256, 0, stream>>>(scores, top_idx, top_val);
  encode_pool_kernel<<<B_, 256, 0, stream>>>(
      seq, pos, top_idx, top_val, wte1, be1, wte2, be2, gam, bet, out);
}
